// Model_34643206210293
// MI455X (gfx1250) — matrix operand order repaired and checked by reading, not compiled, not run on hardware
//
#include <hip/hip_runtime.h>
#include <hip/hip_bf16.h>

typedef __attribute__((ext_vector_type(16))) _Float16 v16h;
typedef __attribute__((ext_vector_type(8)))  _Float16 v8h;
typedef __attribute__((ext_vector_type(8)))  float    v8f;

#define NTAGS 101
#define NPAD  112   // 7 N-tiles of 16
#define KPAD  128   // 4 K-chunks of 32
#define BATCH 256
#define SEQL  2048
#define BG    16    // batch rows per block (WMMA M)

#define LOG2E 1.44269504088896340736f
#define LN2   0.69314718055994530942f

// ---- workspace layout (bytes) ----
#define WS_TRANS_OFF 0                      // float[101*101]
#define WS_TEXP_OFF  40960                  // _Float16[112*128]  ([n][k])
#define WS_FWD_OFF   (40960 + 28672)        // float[256]

__device__ __forceinline__ float fast_exp(float x) {
    return __builtin_amdgcn_exp2f(x * LOG2E);   // v_exp_f32 computes 2^x
}
__device__ __forceinline__ float fast_log(float x) {
    return __builtin_amdgcn_logf(x) * LN2;      // v_log_f32 computes log2(x)
}

// ---------------------------------------------------------------------------
// Kernel 1: build trans (fp32) and exp(trans) in f16, padded, B-layout [n][k]
// ---------------------------------------------------------------------------
__global__ __launch_bounds__(256) void crf_prep_kernel(
    const float* __restrict__ cdt, const int* __restrict__ types0,
    const int* __restrict__ types1, float* __restrict__ trans,
    _Float16* __restrict__ texp)
{
    const int tid = threadIdx.x;
    for (int idx = tid; idx < NTAGS * NTAGS; idx += 256)
        trans[idx] = cdt[types0[idx] * 5 + types1[idx]];
    __syncthreads();
    for (int idx = tid; idx < NPAD * KPAD; idx += 256) {
        const int n = idx >> 7;          // column j of trans
        const int k = idx & 127;         // row i of trans
        float v = 0.f;
        if (n < NTAGS && k < NTAGS) v = fast_exp(trans[k * NTAGS + n]);
        texp[idx] = (_Float16)v;
    }
}

// ---------------------------------------------------------------------------
// Kernel 2: the CRF forward scan. One block = 16 batch rows, 8 waves.
// Waves 0..6 each own one 16-wide N-tile; wave 7 prefetches mask bits.
// ---------------------------------------------------------------------------
__global__ __launch_bounds__(256) void crf_forward_kernel(
    const float* __restrict__ feats, const float* __restrict__ start_t,
    const float* __restrict__ stop_t, const int* __restrict__ mask,
    const _Float16* __restrict__ texp, float* __restrict__ fwd_out)
{
    __shared__ __align__(32) _Float16 Bsm[NPAD][KPAD];  // exp(trans), [n][k]
    __shared__ __align__(32) _Float16 Asm[BG][KPAD];    // exp(p - m), f16
    __shared__ __align__(16) float    Psm[BG][KPAD];    // partition, -1e30 pad
    __shared__ float Msm[BG];                           // per-row max
    __shared__ int   Mkm[2][BG];                        // per-row mask, dbl-buf

    const int tid  = threadIdx.x;
    const int lane = tid & 31;
    const int wave = tid >> 5;
    const int b0   = blockIdx.x * BG;

    // stage constant exp(trans) into LDS
    for (int idx = tid; idx < NPAD * KPAD; idx += 256)
        (&Bsm[0][0])[idx] = texp[idx];

    // init partition: feats[:,0,:] + start_transitions ; pad cols -> -1e30
    for (int idx = tid; idx < BG * KPAD; idx += 256) {
        const int b = idx >> 7, i = idx & 127;
        float v = -1e30f;
        if (i < NTAGS)
            v = feats[((size_t)(b0 + b) * SEQL) * NTAGS + i] + start_t[i];
        Psm[b][i] = v;
    }
    // mask bits for t = 1
    if (tid < BG) Mkm[1][tid] = mask[(size_t)(b0 + tid) * SEQL + 1];
    __syncthreads();

    const int nloc = lane & 15;
    const int j    = wave * 16 + nloc;          // output column this lane owns
    const int Mhi  = (lane >= 16) ? 8 : 0;      // C rows 8..15 in hi lanes
    const int khA  = (lane >= 16) ? 16 : 0;     // A: hi lanes hold K+16..31
    const float* fbase =
        feats + ((size_t)(b0 + Mhi) * SEQL) * NTAGS + (j < NTAGS ? j : 0);

    // hoist constant B fragments into registers (per wave N-tile)
    v16h bfrag[4];
    if (wave < 7) {
        const int khB = (lane >= 16) ? 16 : 0;  // B: lanes 16-31 hold K+16..31
#pragma unroll
        for (int kc = 0; kc < 4; ++kc)
            bfrag[kc] = *(const v16h*)&Bsm[j][kc * 32 + khB];
    }

    // software pipeline: preload feats for t = 1
    float fbuf[8];
    if (wave < 7) {
#pragma unroll
        for (int r = 0; r < 8; ++r)
            fbuf[r] = fbase[(size_t)r * SEQL * NTAGS + (size_t)NTAGS];
    }

    const int rowAB = tid >> 4;   // 16 lanes per partition row
    const int subAB = tid & 15;

    for (int t = 1; t < SEQL; ++t) {
        // ---- phase AB: parallel row-max + exp into Asm (LDS only) ----
        {
            const float4* pr = (const float4*)&Psm[rowAB][subAB * 8];
            const float4 x0 = pr[0], x1 = pr[1];
            float m = fmaxf(fmaxf(fmaxf(x0.x, x0.y), fmaxf(x0.z, x0.w)),
                            fmaxf(fmaxf(x1.x, x1.y), fmaxf(x1.z, x1.w)));
            m = fmaxf(m, __shfl_xor(m, 1, 32));
            m = fmaxf(m, __shfl_xor(m, 2, 32));
            m = fmaxf(m, __shfl_xor(m, 4, 32));
            m = fmaxf(m, __shfl_xor(m, 8, 32));
            const float xs[8] = {x0.x, x0.y, x0.z, x0.w, x1.x, x1.y, x1.z, x1.w};
            union { v8h v; _Float16 e[8]; } pk;
#pragma unroll
            for (int e = 0; e < 8; ++e)
                pk.e[e] = (_Float16)__builtin_amdgcn_exp2f((xs[e] - m) * LOG2E);
            *(v8h*)&Asm[rowAB][subAB * 8] = pk.v;
            if (subAB == 0) Msm[rowAB] = m;
        }
        __syncthreads();

        // ---- phase C: WMMA (2 chains) + branchless fixup + prefetches ----
        if (wave < 7) {
            union { v16h v; v8h h[2]; } a0, a1, a2, a3;
            a0.h[0] = *(const v8h*)&Asm[nloc][0   + khA];
            a0.h[1] = *(const v8h*)&Asm[nloc][8   + khA];
            a1.h[0] = *(const v8h*)&Asm[nloc][32  + khA];
            a1.h[1] = *(const v8h*)&Asm[nloc][40  + khA];
            a2.h[0] = *(const v8h*)&Asm[nloc][64  + khA];
            a2.h[1] = *(const v8h*)&Asm[nloc][72  + khA];
            a3.h[0] = *(const v8h*)&Asm[nloc][96  + khA];
            a3.h[1] = *(const v8h*)&Asm[nloc][104 + khA];
            v8f c0 = {}, c1 = {};
            c0 = __builtin_amdgcn_wmma_f32_16x16x32_f16(
                     false, a0.v, false, bfrag[0], (short)0, c0, false, false);
            c1 = __builtin_amdgcn_wmma_f32_16x16x32_f16(
                     false, a2.v, false, bfrag[2], (short)0, c1, false, false);
            c0 = __builtin_amdgcn_wmma_f32_16x16x32_f16(
                     false, a1.v, false, bfrag[1], (short)0, c0, false, false);
            c1 = __builtin_amdgcn_wmma_f32_16x16x32_f16(
                     false, a3.v, false, bfrag[3], (short)0, c1, false, false);
            const v8f c = c0 + c1;

            // prefetch feats for next step (clamped at the end)
            const int tn = (t + 1 < SEQL) ? (t + 1) : t;
            float fnew[8];
#pragma unroll
            for (int r = 0; r < 8; ++r)
                fnew[r] = fbase[(size_t)r * SEQL * NTAGS + (size_t)tn * NTAGS];

            if (j < NTAGS) {
#pragma unroll
                for (int r = 0; r < 8; ++r) {
                    const int bm = r + Mhi;
                    const float pold = Psm[bm][j];
                    const float upd  = fbuf[r] + Msm[bm]
                                     + __builtin_amdgcn_logf((float)c[r]) * LN2;
                    Psm[bm][j] = Mkm[t & 1][bm] ? upd : pold;
                }
            }
#pragma unroll
            for (int r = 0; r < 8; ++r) fbuf[r] = fnew[r];
        } else {
            // wave 7: prefetch mask bits for step t+1 into the other buffer
            if (lane < BG) {
                const int tnm = (t + 1 < SEQL) ? (t + 1) : t;
                Mkm[(t + 1) & 1][lane] =
                    mask[(size_t)(b0 + lane) * SEQL + tnm];
            }
        }
        __syncthreads();
    }

    // forward_score = logsumexp_j(partition + stop_transitions)
    if (tid < BG) {
        float m = -1e30f;
        for (int jj = 0; jj < NTAGS; ++jj)
            m = fmaxf(m, Psm[tid][jj] + stop_t[jj]);
        float s = 0.f;
        for (int jj = 0; jj < NTAGS; ++jj)
            s += fast_exp(Psm[tid][jj] + stop_t[jj] - m);
        fwd_out[b0 + tid] = m + fast_log(s);
    }
}

// ---------------------------------------------------------------------------
// Kernel 3: gold score + final output. One block per batch row.
// ---------------------------------------------------------------------------
__global__ __launch_bounds__(256) void crf_gold_kernel(
    const float* __restrict__ feats, const float* __restrict__ trans,
    const float* __restrict__ start_t, const float* __restrict__ stop_t,
    const int* __restrict__ mask, const int* __restrict__ tags,
    const float* __restrict__ fwd, float* __restrict__ out)
{
    __shared__ float redf[256];
    __shared__ int   redi[256];
    const int b = blockIdx.x, tid = threadIdx.x;
    const size_t baseT = (size_t)b * SEQL;
    const size_t baseF = baseT * NTAGS;

    float acc = 0.f;
    int   len = 0;
    for (int t = tid; t < SEQL; t += 256) {
        if (mask[baseT + t]) {
            const int tg = tags[baseT + t];
            acc += feats[baseF + (size_t)t * NTAGS + tg];
            if (t > 0) acc += trans[tags[baseT + t - 1] * NTAGS + tg];
            len += 1;
        }
    }
    redf[tid] = acc;
    redi[tid] = len;
    __syncthreads();
    for (int s = 128; s > 0; s >>= 1) {
        if (tid < s) { redf[tid] += redf[tid + s]; redi[tid] += redi[tid + s]; }
        __syncthreads();
    }
    if (tid == 0) {
        const int L  = redi[0];
        const int t0 = tags[baseT];
        const int te = tags[baseT + L - 1];
        const float gold = redf[0] + start_t[t0] + stop_t[te];
        out[b] = fwd[b] - gold;
    }
}

// ---------------------------------------------------------------------------
extern "C" void kernel_launch(void* const* d_in, const int* in_sizes, int n_in,
                              void* d_out, int out_size, void* d_ws, size_t ws_size,
                              hipStream_t stream) {
    const float* feats   = (const float*)d_in[0];
    const float* cdt     = (const float*)d_in[1];
    const float* start_t = (const float*)d_in[2];
    const float* stop_t  = (const float*)d_in[3];
    const int*   mask    = (const int*)d_in[4];
    const int*   tags    = (const int*)d_in[5];
    const int*   types0  = (const int*)d_in[6];
    const int*   types1  = (const int*)d_in[7];
    float*       out     = (float*)d_out;

    char* ws = (char*)d_ws;
    float*     trans = (float*)(ws + WS_TRANS_OFF);
    _Float16*  texp  = (_Float16*)(ws + WS_TEXP_OFF);
    float*     fwd   = (float*)(ws + WS_FWD_OFF);

    crf_prep_kernel<<<1, 256, 0, stream>>>(cdt, types0, types1, trans, texp);
    crf_forward_kernel<<<BATCH / BG, 256, 0, stream>>>(feats, start_t, stop_t,
                                                       mask, texp, fwd);
    crf_gold_kernel<<<BATCH, 256, 0, stream>>>(feats, trans, start_t, stop_t,
                                               mask, tags, fwd, out);
}
